// SwinTransformerBlock1D_16930761081044
// MI455X (gfx1250) — matrix operand order repaired and checked by reading, not compiled, not run on hardware
//
#include <hip/hip_runtime.h>
#include <math.h>

// ---------------- problem constants ----------------
#define BDIM  8
#define LSEQ  16384
#define CDIM  128
#define HHEAD 4
#define WSZ   64
#define SHIFT 32
#define NWIN  (LSEQ / WSZ)   // 256
#define HD    32
#define HID   512
#define SCALE 0.17677669529663687f

typedef __attribute__((ext_vector_type(16))) __bf16 v16bf;
typedef __attribute__((ext_vector_type(8)))  __bf16 v8bf;
typedef __attribute__((ext_vector_type(8)))  float  v8f;

#define WMMA_BF16(A, B, Cacc) \
  __builtin_amdgcn_wmma_f32_16x16x32_bf16(false, (A), false, (B), (short)0, (Cacc), false, false)

// A-fragment (16x32 bf16): lane m = lane%16; k = k0+16h..+15 (contiguous)
__device__ __forceinline__ v16bf load_frag_A(const unsigned short* rowptr, int k0, int h8) {
  union { v16bf v; v8bf h[2]; } u;
  u.h[0] = *(const v8bf*)(rowptr + k0 + 2 * h8);
  u.h[1] = *(const v8bf*)(rowptr + k0 + 2 * h8 + 8);
  return u.v;
}
// B-fragment (32x16 bf16): lane n = lane%16; 16 contiguous halves at koff (incl. half*16)
__device__ __forceinline__ v16bf load_frag_B(const unsigned short* rowptr, int koff) {
  union { v16bf v; v8bf h[2]; } u;
  u.h[0] = *(const v8bf*)(rowptr + koff);
  u.h[1] = *(const v8bf*)(rowptr + koff + 8);
  return u.v;
}
// native-converted bf16 pack of a C-fragment (+per-reg bias) -> one b128 store payload
__device__ __forceinline__ v8bf cvt_frag(const v8f& a, const float* bias8) {
  v8bf p;
#pragma unroll
  for (int i = 0; i < 8; i++) p[i] = (__bf16)(a[i] + bias8[i]);
  return p;
}

// async global -> LDS copy, 16B per lane (GLOBAL_LOAD_ASYNC_TO_LDS_B128, ASYNCcnt)
__device__ __forceinline__ void async_copy_b128(void* lds_dst, const void* gsrc) {
  unsigned lds_off = (unsigned)(unsigned long long)lds_dst;   // LDS generic addr[31:0] = offset
  unsigned long long ga = (unsigned long long)gsrc;
  asm volatile("global_load_async_to_lds_b128 %0, %1, off"
               :: "v"(lds_off), "v"(ga) : "memory");
}
__device__ __forceinline__ void async_wait_all() {
  asm volatile("s_wait_asynccnt 0x0" ::: "memory");
}

// ---------------- kernel 0: weight transpose+bf16, bias gather ----------------
__global__ __launch_bounds__(256) void prep_kernel(
    const float* __restrict__ w_qkv, const float* __restrict__ w_proj,
    const float* __restrict__ w_fc1, const float* __restrict__ w_fc2,
    const float* __restrict__ rel_bias,
    unsigned short* __restrict__ w_qkvT, unsigned short* __restrict__ w_projT,
    unsigned short* __restrict__ w_fc1T, unsigned short* __restrict__ w_fc2T,
    float* __restrict__ biasTab) {
  int tid = blockIdx.x * blockDim.x + threadIdx.x;
  int total = 384 * 128 + 128 * 128 + 512 * 128 + 128 * 512 + HHEAD * WSZ * WSZ;
  for (int i = tid; i < total; i += gridDim.x * blockDim.x) {
    int j = i;
    if (j < 384 * 128) { int n = j / 128, k = j % 128;
      *(__bf16*)&w_qkvT[j] = (__bf16)w_qkv[k * 384 + n]; continue; }
    j -= 384 * 128;
    if (j < 128 * 128) { int n = j / 128, k = j % 128;
      *(__bf16*)&w_projT[j] = (__bf16)w_proj[k * 128 + n]; continue; }
    j -= 128 * 128;
    if (j < 512 * 128) { int n = j / 128, k = j % 128;
      *(__bf16*)&w_fc1T[j] = (__bf16)w_fc1[k * 512 + n]; continue; }
    j -= 512 * 128;
    if (j < 128 * 512) { int n = j / 512, k = j % 512;
      *(__bf16*)&w_fc2T[j] = (__bf16)w_fc2[k * 128 + n]; continue; }
    j -= 128 * 512;
    { int h = j / (WSZ * WSZ), r = j % (WSZ * WSZ); int q = r / WSZ, k = r % WSZ;
      biasTab[j] = rel_bias[(q - k + WSZ - 1) * HHEAD + h]; }
  }
}

// ---------------- kernel 1: LN1 + shifted-window attention + proj + residual ----------------
__global__ __launch_bounds__(256) void attn_kernel(
    const float* __restrict__ x, const float* __restrict__ mask,
    const float* __restrict__ gamma1, const float* __restrict__ beta1,
    const float* __restrict__ b_qkv, const float* __restrict__ b_proj,
    const unsigned short* __restrict__ w_qkvT,
    const unsigned short* __restrict__ w_projT,
    const float* __restrict__ biasTab,
    float* __restrict__ out) {
  __shared__ __align__(16) union {
    float xstage[64][128];                                           // 32KB (phase 0/LN only)
    struct { unsigned short ks[64][128]; unsigned short vT[128][64]; } kv;
  } u1;
  __shared__ __align__(16) unsigned short qs[64][128];               // 16KB, reused as attn-out
  __shared__ __align__(16) union {
    unsigned short xln[64][128];                                     // LN1 output (bf16)
    unsigned short pbuf[8][16][64];                                  // per-wave softmax tile
  } u2;

  const int tid  = threadIdx.x;
  const int lane = tid & 31;
  const int wv   = __builtin_amdgcn_readfirstlane(tid >> 5);  // wave 0..7 (scalar)
  const int half = lane >> 4;                                 // 0/1
  const int ln16 = lane & 15;
  const int h8   = half * 8;
  const int blk  = blockIdx.x;                                // b*NWIN + win
  const int b    = blk / NWIN;
  const int win  = blk % NWIN;
  const size_t xbase = (size_t)b * LSEQ * CDIM;

  // phase 0: async DMA of shifted window rows into LDS (roll -SHIFT)
  for (int i = 0; i < 8; i++) {
    int idx = tid + i * 256;                 // 2048 x 16B
    int g = idx >> 5;                        // row in window
    int cc = (idx & 31) << 2;                // column (float4 granular)
    int row = (win * WSZ + g + SHIFT) & (LSEQ - 1);
    async_copy_b128(&u1.xstage[g][cc], x + xbase + (size_t)row * CDIM + cc);
  }
  async_wait_all();
  __syncthreads();

  // LN1: 4 lanes per token, shuffle reduce, native bf16 packed stores
  {
    int tok = tid >> 2, q = tid & 3;
    float s = 0.f, s2 = 0.f;
    for (int i = 0; i < 8; i++) {
      float4 v = *(const float4*)&u1.xstage[tok][q * 32 + i * 4];
      s  += v.x + v.y + v.z + v.w;
      s2 += v.x * v.x + v.y * v.y + v.z * v.z + v.w * v.w;
    }
    s += __shfl_xor(s, 1); s2 += __shfl_xor(s2, 1);
    s += __shfl_xor(s, 2); s2 += __shfl_xor(s2, 2);
    float mean = s * (1.0f / 128.0f);
    float var  = s2 * (1.0f / 128.0f) - mean * mean;
    float rstd = rsqrtf(var + 1e-5f);
    for (int g4 = 0; g4 < 4; g4++) {
      int ch = q * 32 + g4 * 8;
      union { float4 v4[2]; float f[8]; } ga, be, xv;
      ga.v4[0] = *(const float4*)&gamma1[ch]; ga.v4[1] = *(const float4*)&gamma1[ch + 4];
      be.v4[0] = *(const float4*)&beta1[ch];  be.v4[1] = *(const float4*)&beta1[ch + 4];
      xv.v4[0] = *(const float4*)&u1.xstage[tok][ch];
      xv.v4[1] = *(const float4*)&u1.xstage[tok][ch + 4];
      v8bf p;
#pragma unroll
      for (int i = 0; i < 8; i++)
        p[i] = (__bf16)((xv.f[i] - mean) * rstd * ga.f[i] + be.f[i]);
      *(v8bf*)&u2.xln[tok][ch] = p;
    }
  }
  __syncthreads();

  // QKV GEMM: 24 nt tiles over 8 waves (3 each); weight fragments hoisted across 4 mt tiles.
  for (int ni = 0; ni < 3; ni++) {
    int nt = wv * 3 + ni;                          // scalar (wv is scalar)
    const unsigned short* wrow = w_qkvT + (size_t)(nt * 16 + ln16) * 128;
    if (nt < 16) {
      // Q/K: Dt = Wtile x xln^T : rows = out-ch, cols = token
      v16bf wa[4];
      for (int kk = 0; kk < 4; kk++) wa[kk] = load_frag_A(wrow, kk * 32, h8);
      int chb = nt * 16 + h8;                      // 8 consecutive out-channels
      union { float4 v4[2]; float f[8]; } bb;
      bb.v4[0] = *(const float4*)&b_qkv[chb];
      bb.v4[1] = *(const float4*)&b_qkv[chb + 4];
      for (int mt = 0; mt < 4; mt++) {
        v8f acc = {};
        const unsigned short* brow = &u2.xln[mt * 16 + ln16][0];
        for (int kk = 0; kk < 4; kk++)
          acc = WMMA_BF16(wa[kk], load_frag_B(brow, kk * 32 + half * 16), acc);
        int tok = mt * 16 + ln16;
        v8bf p = cvt_frag(acc, bb.f);
        unsigned short* dst = (nt < 8) ? &qs[tok][chb] : &u1.kv.ks[tok][chb - 128];
        *(v8bf*)dst = p;
      }
    } else {
      // V: normal orientation, store d-major (rows = token -> contiguous in vT row)
      v16bf wb[4];
      for (int kk = 0; kk < 4; kk++) wb[kk] = load_frag_B(wrow, kk * 32 + half * 16);
      int ncol = nt * 16 + ln16;                   // 256..383
      float bias = b_qkv[ncol];
      float bbv[8] = {bias, bias, bias, bias, bias, bias, bias, bias};
      for (int mt = 0; mt < 4; mt++) {
        v8f acc = {};
        const unsigned short* arow = &u2.xln[mt * 16 + ln16][0];
        for (int kk = 0; kk < 4; kk++)
          acc = WMMA_BF16(load_frag_A(arow, kk * 32, h8), wb[kk], acc);
        v8bf p = cvt_frag(acc, bbv);
        *(v8bf*)&u1.kv.vT[ncol - 256][mt * 16 + h8] = p;
      }
    }
  }
  __syncthreads();

  // attention (transposed): attn^T = K x Q^T. lane col = query token, regs = keys.
  v8f outf[2][2];
  for (int ti = 0; ti < 2; ti++) {
    int task = wv + ti * 8;
    int head = task >> 2, mt = task & 3;
    v8f cfr[4];
    {
      const unsigned short* brow = &qs[mt * 16 + ln16][0];
      v16bf bq = load_frag_B(brow, head * 32 + half * 16);
      for (int ct = 0; ct < 4; ct++) {
        const unsigned short* arow = &u1.kv.ks[ct * 16 + ln16][0];
        v16bf ak = load_frag_A(arow, head * 32, h8);
        v8f z = {};
        cfr[ct] = WMMA_BF16(ak, bq, z);
      }
    }
    // logits: query = mt*16+ln16 (per lane), key = ct*16 + r + half*8 (contiguous per frag)
    int qrow = mt * 16 + ln16;
    const float* bt = biasTab + (size_t)head * WSZ * WSZ + qrow * WSZ;
    const float* mr = mask + (size_t)win * WSZ * WSZ + qrow * WSZ;
    for (int ct = 0; ct < 4; ct++) {
      int kb = ct * 16 + h8;
      union { float4 v4[2]; float f[8]; } bbu, mmu;
      bbu.v4[0] = *(const float4*)(bt + kb); bbu.v4[1] = *(const float4*)(bt + kb + 4);
      mmu.v4[0] = *(const float4*)(mr + kb); mmu.v4[1] = *(const float4*)(mr + kb + 4);
#pragma unroll
      for (int r = 0; r < 8; r++)
        cfr[ct][r] = cfr[ct][r] * SCALE + bbu.f[r] + mmu.f[r];
    }
    // softmax over 64 keys: 32 in-lane values + one xor-16 shuffle to combine halves
    float mx = cfr[0][0];
    for (int ct = 0; ct < 4; ct++)
      for (int r = 0; r < 8; r++) mx = fmaxf(mx, cfr[ct][r]);
    mx = fmaxf(mx, __shfl_xor(mx, 16));
    float sum = 0.f;
    for (int ct = 0; ct < 4; ct++)
      for (int r = 0; r < 8; r++) { float e = __expf(cfr[ct][r] - mx); cfr[ct][r] = e; sum += e; }
    sum += __shfl_xor(sum, 16);
    float inv = 1.0f / sum;
    // P^T into pbuf: one packed b128 per frag (keys contiguous per lane)
    for (int ct = 0; ct < 4; ct++) {
      v8bf p;
#pragma unroll
      for (int i = 0; i < 8; i++) p[i] = (__bf16)(cfr[ct][i] * inv);
      *(v8bf*)&u2.pbuf[wv][ln16][ct * 16 + h8] = p;
    }
    // out^T = V^T x P^T : rows = d, cols = token. K = 64 keys = 2 steps.
    for (int dt = 0; dt < 2; dt++) {
      v8f acc = {};
      const unsigned short* arow = &u1.kv.vT[head * 32 + dt * 16 + ln16][0];
      const unsigned short* brow = &u2.pbuf[wv][ln16][0];
      for (int k2 = 0; k2 < 2; k2++) {
        v16bf av = load_frag_A(arow, k2 * 32, h8);
        v16bf bp = load_frag_B(brow, k2 * 32 + half * 16);
        acc = WMMA_BF16(av, bp, acc);
      }
      outf[ti][dt] = acc;
    }
  }
  __syncthreads();

  // scatter O (held in regs across barrier) into qs region, token-major, packed b128
  unsigned short (*os)[128] = qs;
  const float zbias[8] = {0, 0, 0, 0, 0, 0, 0, 0};
  for (int ti = 0; ti < 2; ti++) {
    int task = wv + ti * 8;
    int head = task >> 2, mt = task & 3;
    int tok = mt * 16 + ln16;
    for (int dt = 0; dt < 2; dt++) {
      v8bf p = cvt_frag(outf[ti][dt], zbias);
      *(v8bf*)&os[tok][head * 32 + dt * 16 + h8] = p;
    }
  }
  __syncthreads();

  // proj (transposed): wave -> fixed mt; hoist os B-fragments over the 4 nt weight tiles
  {
    int mt = wv >> 1;
    int nbase = (wv & 1) * 4;
    const unsigned short* brow = &os[mt * 16 + ln16][0];
    v16bf ob[4];
    for (int kk = 0; kk < 4; kk++) ob[kk] = load_frag_B(brow, kk * 32 + half * 16);
    int tok = mt * 16 + ln16;
    int row = (win * WSZ + tok + SHIFT) & (LSEQ - 1);
    for (int t = 0; t < 4; t++) {
      int nt = nbase + t;
      v8f acc = {};
      const unsigned short* arow = w_projT + (size_t)(nt * 16 + ln16) * 128;
      for (int kk = 0; kk < 4; kk++)
        acc = WMMA_BF16(load_frag_A(arow, kk * 32, h8), ob[kk], acc);
      int chb = nt * 16 + h8;
      const float* xp = x + xbase + (size_t)row * CDIM + chb;
      float*       op = out + xbase + (size_t)row * CDIM + chb;
      float4 bb0 = *(const float4*)&b_proj[chb];
      float4 bb1 = *(const float4*)&b_proj[chb + 4];
      float4 x0 = *(const float4*)xp;
      float4 x1 = *(const float4*)(xp + 4);
      float4 o0, o1;
      o0.x = x0.x + acc[0] + bb0.x; o0.y = x0.y + acc[1] + bb0.y;
      o0.z = x0.z + acc[2] + bb0.z; o0.w = x0.w + acc[3] + bb0.w;
      o1.x = x1.x + acc[4] + bb1.x; o1.y = x1.y + acc[5] + bb1.y;
      o1.z = x1.z + acc[6] + bb1.z; o1.w = x1.w + acc[7] + bb1.w;
      *(float4*)op = o0;
      *(float4*)(op + 4) = o1;
    }
  }
}

// ---------------- kernel 2: LN2 + MLP + residual (in-place on out = x1) ----------------
__global__ __launch_bounds__(256) void mlp_kernel(
    const float* __restrict__ gamma2, const float* __restrict__ beta2,
    const float* __restrict__ b_fc1, const float* __restrict__ b_fc2,
    const unsigned short* __restrict__ w_fc1T,
    const unsigned short* __restrict__ w_fc2T,
    float* __restrict__ out) {
  __shared__ __align__(16) unsigned short xln[32][128];   // 8KB
  __shared__ __align__(16) union {
    float stage[32][128];                                 // 16KB (LN staging)
    unsigned short hbuf[32][512];                         // 32KB (GELU(fc1) bf16)
  } uu;

  const int tid = threadIdx.x, lane = tid & 31;
  const int wv = __builtin_amdgcn_readfirstlane(tid >> 5);
  const int half = lane >> 4, ln16 = lane & 15, h8 = half * 8;
  const size_t base = (size_t)blockIdx.x * 32 * CDIM;

  // async DMA of x1 tile into LDS
  for (int i = 0; i < 4; i++) {
    int idx = tid + i * 256;
    int g = idx >> 5; int cc = (idx & 31) << 2;
    async_copy_b128(&uu.stage[g][cc], out + base + (size_t)g * CDIM + cc);
  }
  async_wait_all();
  __syncthreads();
  // LN2: 8 lanes per token, native bf16 packed stores
  {
    int tok = tid >> 3, q = tid & 7;
    float s = 0.f, s2 = 0.f;
    for (int i = 0; i < 4; i++) {
      float4 v = *(const float4*)&uu.stage[tok][q * 16 + i * 4];
      s  += v.x + v.y + v.z + v.w;
      s2 += v.x * v.x + v.y * v.y + v.z * v.z + v.w * v.w;
    }
    s += __shfl_xor(s, 1); s2 += __shfl_xor(s2, 1);
    s += __shfl_xor(s, 2); s2 += __shfl_xor(s2, 2);
    s += __shfl_xor(s, 4); s2 += __shfl_xor(s2, 4);
    float mean = s * (1.0f / 128.0f);
    float var  = s2 * (1.0f / 128.0f) - mean * mean;
    float rstd = rsqrtf(var + 1e-5f);
    for (int g2 = 0; g2 < 2; g2++) {
      int ch = q * 16 + g2 * 8;
      union { float4 v4[2]; float f[8]; } ga, be, xv;
      ga.v4[0] = *(const float4*)&gamma2[ch]; ga.v4[1] = *(const float4*)&gamma2[ch + 4];
      be.v4[0] = *(const float4*)&beta2[ch];  be.v4[1] = *(const float4*)&beta2[ch + 4];
      xv.v4[0] = *(const float4*)&uu.stage[tok][ch];
      xv.v4[1] = *(const float4*)&uu.stage[tok][ch + 4];
      v8bf p;
#pragma unroll
      for (int i = 0; i < 8; i++)
        p[i] = (__bf16)((xv.f[i] - mean) * rstd * ga.f[i] + be.f[i]);
      *(v8bf*)&xln[tok][ch] = p;
    }
  }
  __syncthreads();
  // fc1 (transposed) + exact GELU -> hbuf; weight fragments hoisted over mt
  for (int ni = 0; ni < 4; ni++) {
    int nt = wv * 4 + ni;                       // hid tile 0..31
    const unsigned short* arow = w_fc1T + (size_t)(nt * 16 + ln16) * 128;
    v16bf wa[4];
    for (int kk = 0; kk < 4; kk++) wa[kk] = load_frag_A(arow, kk * 32, h8);
    int chb = nt * 16 + h8;
    union { float4 v4[2]; float f[8]; } bb;
    bb.v4[0] = *(const float4*)&b_fc1[chb];
    bb.v4[1] = *(const float4*)&b_fc1[chb + 4];
    for (int mt = 0; mt < 2; mt++) {
      v8f acc = {};
      const unsigned short* brow = &xln[mt * 16 + ln16][0];
      for (int kk = 0; kk < 4; kk++)
        acc = WMMA_BF16(wa[kk], load_frag_B(brow, kk * 32 + half * 16), acc);
      int tok = mt * 16 + ln16;
      v8bf p;
#pragma unroll
      for (int r = 0; r < 8; r++) {
        float v = acc[r] + bb.f[r];
        p[r] = (__bf16)(0.5f * v * (1.0f + erff(v * 0.70710678118654752f)));
      }
      *(v8bf*)&uu.hbuf[tok][chb] = p;
    }
  }
  __syncthreads();
  // fc2 (transposed) + residual: contiguous float4 read-modify-write
  for (int t = 0; t < 2; t++) {
    int tt = wv * 2 + t;
    int mt = tt >> 3, nt = tt & 7;
    v8f acc = {};
    const unsigned short* arow = w_fc2T + (size_t)(nt * 16 + ln16) * 512;
    const unsigned short* brow = &uu.hbuf[mt * 16 + ln16][0];
    for (int kk = 0; kk < 16; kk++) {
      v16bf a  = load_frag_A(arow, kk * 32, h8);
      v16bf bm = load_frag_B(brow, kk * 32 + half * 16);
      acc = WMMA_BF16(a, bm, acc);
    }
    int tok = mt * 16 + ln16;
    int chb = nt * 16 + h8;
    float* op = out + base + (size_t)tok * CDIM + chb;
    float4 bb0 = *(const float4*)&b_fc2[chb];
    float4 bb1 = *(const float4*)&b_fc2[chb + 4];
    float4 x0 = *(const float4*)op;
    float4 x1 = *(const float4*)(op + 4);
    float4 o0, o1;
    o0.x = x0.x + acc[0] + bb0.x; o0.y = x0.y + acc[1] + bb0.y;
    o0.z = x0.z + acc[2] + bb0.z; o0.w = x0.w + acc[3] + bb0.w;
    o1.x = x1.x + acc[4] + bb1.x; o1.y = x1.y + acc[5] + bb1.y;
    o1.z = x1.z + acc[6] + bb1.z; o1.w = x1.w + acc[7] + bb1.w;
    *(float4*)op = o0;
    *(float4*)(op + 4) = o1;
  }
}

// ---------------- host launcher ----------------
extern "C" void kernel_launch(void* const* d_in, const int* in_sizes, int n_in,
                              void* d_out, int out_size, void* d_ws, size_t ws_size,
                              hipStream_t stream) {
  const float* x        = (const float*)d_in[0];
  const float* mask     = (const float*)d_in[1];
  const float* gamma1   = (const float*)d_in[2];
  const float* beta1    = (const float*)d_in[3];
  const float* w_qkv    = (const float*)d_in[4];
  const float* b_qkv    = (const float*)d_in[5];
  const float* rel_bias = (const float*)d_in[6];
  const float* w_proj   = (const float*)d_in[7];
  const float* b_proj   = (const float*)d_in[8];
  const float* gamma2   = (const float*)d_in[9];
  const float* beta2    = (const float*)d_in[10];
  const float* w_fc1    = (const float*)d_in[11];
  const float* b_fc1    = (const float*)d_in[12];
  const float* w_fc2    = (const float*)d_in[13];
  const float* b_fc2    = (const float*)d_in[14];
  float* out = (float*)d_out;

  char* ws = (char*)d_ws;
  unsigned short* w_qkvT = (unsigned short*)ws; ws += 384 * 128 * 2;
  unsigned short* w_projT = (unsigned short*)ws; ws += 128 * 128 * 2;
  unsigned short* w_fc1T = (unsigned short*)ws; ws += 512 * 128 * 2;
  unsigned short* w_fc2T = (unsigned short*)ws; ws += 128 * 512 * 2;
  float* biasTab = (float*)ws;

  prep_kernel<<<128, 256, 0, stream>>>(w_qkv, w_proj, w_fc1, w_fc2, rel_bias,
                                       w_qkvT, w_projT, w_fc1T, w_fc2T, biasTab);
  attn_kernel<<<BDIM * NWIN, 256, 0, stream>>>(x, mask, gamma1, beta1, b_qkv, b_proj,
                                               w_qkvT, w_projT, biasTab, out);
  mlp_kernel<<<(BDIM * LSEQ) / 32, 256, 0, stream>>>(gamma2, beta2, b_fc1, b_fc2,
                                                     w_fc1T, w_fc2T, out);
}
